// mLSTMCell_14980845928802
// MI455X (gfx1250) — matrix operand order repaired and checked by reading, not compiled, not run on hardware
//
#include <hip/hip_runtime.h>
#include <hip/hip_bf16.h>
#include <math.h>

// mLSTM parallel-stabilized cell, flash-tiled for gfx1250 (MI455X).
// B=2, S=2048, H=1024, NH=4, DH=256.
#define B_  2
#define S_  2048
#define H_  1024
#define NH_ 4
#define DH_ 256

#define DEV_INLINE __device__ __forceinline__

typedef __attribute__((ext_vector_type(2)))  __bf16   v2bf;
typedef __attribute__((ext_vector_type(8)))  __bf16   v8bf;
typedef __attribute__((ext_vector_type(16))) __bf16   v16bf;
typedef __attribute__((ext_vector_type(8)))  float    v8f;
typedef __attribute__((ext_vector_type(4)))  int      v4i;
typedef __attribute__((ext_vector_type(4)))  unsigned v4u;
typedef __attribute__((ext_vector_type(8)))  unsigned v8u;

#if __has_builtin(__builtin_amdgcn_global_load_async_to_lds_b128)
#define HAVE_ASYNC_LDS 1
#else
#define HAVE_ASYNC_LDS 0
#endif

typedef __attribute__((address_space(1))) v4i as1_v4i;
typedef __attribute__((address_space(3))) v4i as3_v4i;

DEV_INLINE as1_v4i* gptr_v4(const void* p) {
  return (as1_v4i*)(unsigned long long)(p);
}
DEV_INLINE as3_v4i* lptr_v4(const void* p) {
  return (as3_v4i*)(unsigned int)(unsigned long long)(p);
}

DEV_INLINE void wait_async0() {
#if __has_builtin(__builtin_amdgcn_s_wait_asynccnt)
  __builtin_amdgcn_s_wait_asynccnt(0);
#else
  asm volatile("s_wait_asynccnt 0x0" ::: "memory");
#endif
}
DEV_INLINE void wait_tensor0() {
#if __has_builtin(__builtin_amdgcn_s_wait_tensorcnt)
  __builtin_amdgcn_s_wait_tensorcnt(0);
#else
  asm volatile("s_wait_tensorcnt 0x0" ::: "memory");
#endif
}
DEV_INLINE void wait_ds0() {
#if __has_builtin(__builtin_amdgcn_s_wait_dscnt)
  __builtin_amdgcn_s_wait_dscnt(0);
#else
  asm volatile("s_wait_dscnt 0x0" ::: "memory");
#endif
}

// Tensor Data Mover: 2D tile load, global -> LDS.  2-group descriptor form
// (VADDR2/VADDR3 = NULL, tensors up to 2D).  All descriptor fields per
// CDNA5 ISA.
// rows x 256 f32 tile, row stride H_ elements, dense into LDS at lds_byte.
DEV_INLINE void tdm_load_rows_f32(const float* gsrc, unsigned lds_byte,
                                  unsigned rows) {
  const unsigned long long ga = (unsigned long long)gsrc;
  const v4u g0 = {
      1u,                                   // count=1, user desc, no gather
      lds_byte,                             // lds_addr [63:32]
      (unsigned)ga,                         // global_addr [95:64]
      ((unsigned)(ga >> 32) & 0x01ffffffu)  // global_addr [120:96]
          | (2u << 30)                      // type=2 ("image") [127:126]
  };
  const v8u g1 = {
      2u << 16,                             // data_size=2 (4B); no multicast
      (DH_ & 0xffffu) << 16,                // tensor_dim0[15:0] @ [63:48]
      (rows & 0xffffu) << 16,               // tensor_dim1[15:0] @ [95:80]
      (DH_ & 0xffffu) << 16,                // tile_dim0 @ [127:112]
      rows & 0xffffu,                       // tile_dim1 @ [143:128]
      (unsigned)H_,                         // tensor_dim0_stride[31:0]
      0u,                                   // dim0_stride hi / dim1_stride lo
      0u                                    // tensor_dim1_stride hi (unused 2D)
  };
  asm volatile("tensor_load_to_lds %0, %1" :: "s"(g0), "s"(g1) : "memory");
}

DEV_INLINE unsigned pack_bf16(float x, float y) {
#if __has_builtin(__builtin_amdgcn_cvt_pk_bf16_f32)
  v2bf p = __builtin_amdgcn_cvt_pk_bf16_f32(x, y);
  return __builtin_bit_cast(unsigned, p);
#else
  union { __bf16 h[2]; unsigned u; } u;
  u.h[0] = (__bf16)x; u.h[1] = (__bf16)y;
  return u.u;
#endif
}

DEV_INLINE v16bf cat16(v8bf lo, v8bf hi) {
  return __builtin_shufflevector(lo, hi, 0, 1, 2, 3, 4, 5, 6, 7,
                                         8, 9, 10, 11, 12, 13, 14, 15);
}

DEV_INLINE v8f wmma_bf16(v16bf a, v16bf b, v8f c) {
  // v_wmma_f32_16x16x32_bf16 : D = A(16x32) * B(32x16) + C(16x16 f32)
  return __builtin_amdgcn_wmma_f32_16x16x32_bf16(
      /*neg_a=*/false, a, /*neg_b=*/false, b,
      /*c_mod=*/(short)0, c, /*reuse_a=*/false, /*reuse_b=*/false);
}

// ---------------------------------------------------------------------------
// Kernel 1: gate projections.  One block per (b,s); ig/fg = x @ W + bias with
// x = concat(q,k,v).  Writes ig and log_sigmoid(fg) in (b,nh,s) layout.
// ---------------------------------------------------------------------------
__global__ void gates_kernel(const float* __restrict__ q,
                             const float* __restrict__ k,
                             const float* __restrict__ v,
                             const float* __restrict__ igk,
                             const float* __restrict__ igb,
                             const float* __restrict__ fgk,
                             const float* __restrict__ fgb,
                             float* __restrict__ ig_out,
                             float* __restrict__ cslog_out) {
  const int bs  = blockIdx.x;           // 0 .. B*S-1
  const int b   = bs / S_;
  const int s   = bs % S_;
  const int tid = threadIdx.x;          // 256 threads

  const float* qp = q + (size_t)bs * H_;
  const float* kp = k + (size_t)bs * H_;
  const float* vp = v + (size_t)bs * H_;

  float accI[NH_] = {0.f, 0.f, 0.f, 0.f};
  float accF[NH_] = {0.f, 0.f, 0.f, 0.f};
  for (int i = tid; i < 3 * H_; i += 256) {
    float x = (i < H_) ? qp[i] : ((i < 2 * H_) ? kp[i - H_] : vp[i - 2 * H_]);
    const float4 wi = *(const float4*)(igk + (size_t)i * NH_);
    const float4 wf = *(const float4*)(fgk + (size_t)i * NH_);
    accI[0] += x * wi.x; accI[1] += x * wi.y; accI[2] += x * wi.z; accI[3] += x * wi.w;
    accF[0] += x * wf.x; accF[1] += x * wf.y; accF[2] += x * wf.z; accF[3] += x * wf.w;
  }

  __shared__ float red[8][256];
#pragma unroll
  for (int h = 0; h < NH_; ++h) {
    red[h][tid]     = accI[h];
    red[h + 4][tid] = accF[h];
  }
  __syncthreads();
  for (int st = 128; st > 0; st >>= 1) {
    if (tid < st) {
#pragma unroll
      for (int h = 0; h < 8; ++h) red[h][tid] += red[h][tid + st];
    }
    __syncthreads();
  }
  if (tid < NH_) {
    const int h    = tid;
    const float iv = red[h][0] + igb[h];
    const float fv = red[h + 4][0] + fgb[h];
    const float lf = fminf(fv, 0.f) - log1pf(expf(-fabsf(fv)));  // log-sigmoid
    const size_t o = ((size_t)(b * NH_ + h)) * S_ + s;
    ig_out[o]    = iv;
    cslog_out[o] = lf;
  }
}

// ---------------------------------------------------------------------------
// Kernel 2: inclusive cumsum of log-forget gates per (b,nh).
// ---------------------------------------------------------------------------
__global__ void scan_kernel(float* __restrict__ cs) {
  if (threadIdx.x != 0) return;
  float* p = cs + (size_t)blockIdx.x * S_;
  float a = 0.f;
  for (int i = 0; i < S_; ++i) { a += p[i]; p[i] = a; }
}

// ---------------------------------------------------------------------------
// Kernel 3: flash mLSTM.  One wave (32 threads) per 16-row tile of one head.
// Online-stabilized over 32-column j-blocks; WMMA bf16 for QK^T and C*V.
// K f32 tiles staged via the Tensor Data Mover, V tiles via async
// global->LDS DMA (both engines run concurrently), then converted to bf16
// in the WMMA operand layouts.
// ---------------------------------------------------------------------------
__global__ void __launch_bounds__(32)
flash_kernel(const float* __restrict__ q,
             const float* __restrict__ k,
             const float* __restrict__ v,
             const float* __restrict__ ig,
             const float* __restrict__ cs,
             const float* __restrict__ onscale,
             float* __restrict__ out) {
  __shared__ __align__(16) __bf16 Qbf[16 * DH_];   //  8 KB, A-source (row major)
  __shared__ __align__(16) __bf16 Kbf[32 * DH_];   // 16 KB, B-source (row major)
  __shared__ __align__(16) __bf16 Vt [DH_ * 32];   // 16 KB, B-source (transposed)
  __shared__ __align__(16) __bf16 Cbf[16 * 32];    //  1 KB, A-staging for PV
  __shared__ __align__(16) float  SfK[8 * DH_];    //  8 KB, TDM f32 staging (K)
  __shared__ __align__(16) float  SfV[8 * DH_];    //  8 KB, async f32 staging (V)

  const int tile  = blockIdx.x;          // 0 .. B*NH*(S/16)-1
  const int rt    = tile & 127;          // row tile within head
  const int bh    = tile >> 7;           // (b*NH + nh)
  const int b     = bh >> 2;
  const int nh    = bh & 3;
  const int rbase = rt * 16;

  const int lane = threadIdx.x;          // wave32, all lanes active
  const int half = lane >> 4;            // 0: lanes 0-15, 1: lanes 16-31
  const int nn   = lane & 15;

  const size_t head_off = (size_t)b * S_ * H_ + (size_t)nh * DH_;

  // ---- stage Q tile, folding 1/sqrt(DH)=1/16 into bf16 conversion ----------
#pragma unroll
  for (int it = 0; it < 32; ++it) {
    const int f4  = it * 32 + lane;                 // 1024 float4 groups
    const int row = f4 >> 6;
    const int c4  = (f4 & 63) * 4;
    const float4 val =
        *(const float4*)(q + head_off + (size_t)(rbase + row) * H_ + c4);
    const int o = row * DH_ + c4;
    *(unsigned*)&Qbf[o]     = pack_bf16(val.x * 0.0625f, val.y * 0.0625f);
    *(unsigned*)&Qbf[o + 2] = pack_bf16(val.z * 0.0625f, val.w * 0.0625f);
  }
  __builtin_amdgcn_wave_barrier();

  // ---- per-row gate state --------------------------------------------------
  const float* csp = cs + (size_t)bh * S_;
  const float* igp = ig + (size_t)bh * S_;
  float csq[8];
#pragma unroll
  for (int r = 0; r < 8; ++r) csq[r] = csp[rbase + r + 8 * half];

  float mrun[8], lrun[8];
  v8f acc[16];
  const v8f zero8 = {0.f, 0.f, 0.f, 0.f, 0.f, 0.f, 0.f, 0.f};
#pragma unroll
  for (int r = 0; r < 8; ++r) { mrun[r] = -1e30f; lrun[r] = 0.f; }
#pragma unroll
  for (int t = 0; t < 16; ++t) acc[t] = zero8;

  // LDS operand base pointers (per-lane, loop-invariant)
  const __bf16* qap  = &Qbf[nn * DH_ + 8 * half];          // A: Q 16x32 chunks
  const __bf16* kbp0 = &Kbf[nn * DH_ + 8 * half];          // B: K^T cols 0-15
  const __bf16* kbp1 = &Kbf[(16 + nn) * DH_ + 8 * half];   // B: K^T cols 16-31
  const __bf16* vtp  = &Vt[nn * 32 + 8 * half];            // B: V 32x16 chunks
  const __bf16* pc   = &Cbf[nn * 32 + 8 * half];           // A: C 16x32

  const unsigned ldsK = (unsigned)(unsigned long long)&SfK[0];

  const int njb = (rbase + 15) / 32 + 1;   // causal j-block count

  for (int jb = 0; jb < njb; ++jb) {
    const int j0 = jb * 32;

    // ---- stage K (row major, TDM) and V (transposed, async) as bf16 -------
#if HAVE_ASYNC_LDS
    for (int part = 0; part < 4; ++part) {     // 8 rows of K and of V per pass
      const int r0 = part * 8;
      // K: one Tensor-DMA descriptor moves the whole 8x256 f32 tile.
      tdm_load_rows_f32(k + head_off + (size_t)(j0 + r0) * H_, ldsK, 8);
      // V: async per-lane DMA, runs concurrently on ASYNCcnt.
#pragma unroll
      for (int it = 0; it < 16; ++it) {
        const int idx = it * 32 + lane;               // 512 float4 slots
        const int row = idx >> 6;                     // 0..7
        const int c4  = (idx & 63) * 4;
        __builtin_amdgcn_global_load_async_to_lds_b128(
            gptr_v4(v + head_off + (size_t)(j0 + r0 + row) * H_ + c4),
            lptr_v4(&SfV[row * DH_ + c4]), 0, 0);
      }
      wait_tensor0();
      wait_async0();
      __builtin_amdgcn_wave_barrier();
      // convert both staged chunks to bf16 operand layouts
#pragma unroll
      for (int it = 0; it < 16; ++it) {
        const int idx = it * 32 + lane;
        const int row = idx >> 6;
        const int c4  = (idx & 63) * 4;
        const float4 kv = *(const float4*)&SfK[row * DH_ + c4];
        const int o = (r0 + row) * DH_ + c4;
        *(unsigned*)&Kbf[o]     = pack_bf16(kv.x, kv.y);
        *(unsigned*)&Kbf[o + 2] = pack_bf16(kv.z, kv.w);
        const float4 vv = *(const float4*)&SfV[row * DH_ + c4];
        const int rr = r0 + row;
        Vt[(c4 + 0) * 32 + rr] = (__bf16)vv.x;
        Vt[(c4 + 1) * 32 + rr] = (__bf16)vv.y;
        Vt[(c4 + 2) * 32 + rr] = (__bf16)vv.z;
        Vt[(c4 + 3) * 32 + rr] = (__bf16)vv.w;
      }
      wait_ds0();                      // staging reads done before next DMA
      __builtin_amdgcn_wave_barrier();
    }
#else
#pragma unroll 4
    for (int it = 0; it < 64; ++it) {
      const int f4  = it * 32 + lane;               // 2048 float4 groups
      const int row = f4 >> 6;                      // 0..31
      const int c4  = (f4 & 63) * 4;                // 0..252
      const size_t g = head_off + (size_t)(j0 + row) * H_ + c4;
      const float4 kv = *(const float4*)(k + g);
      const int o = row * DH_ + c4;
      *(unsigned*)&Kbf[o]     = pack_bf16(kv.x, kv.y);
      *(unsigned*)&Kbf[o + 2] = pack_bf16(kv.z, kv.w);
      const float4 vv = *(const float4*)(v + g);
      Vt[(c4 + 0) * 32 + row] = (__bf16)vv.x;
      Vt[(c4 + 1) * 32 + row] = (__bf16)vv.y;
      Vt[(c4 + 2) * 32 + row] = (__bf16)vv.z;
      Vt[(c4 + 3) * 32 + row] = (__bf16)vv.w;
    }
    __builtin_amdgcn_wave_barrier();
#endif

    // prefetch next j-block of K/V into cache (global_prefetch_b8)
    if (jb + 1 < njb) {
      __builtin_prefetch(k + head_off + (size_t)(j0 + 32) * H_ + lane * 8, 0, 1);
      __builtin_prefetch(v + head_off + (size_t)(j0 + 32) * H_ + lane * 8, 0, 1);
    }

    // ---- S = (Q/sqrt(DH)) * K^T : two 16x16 f32 tiles over DH=256 ----------
    // Software-pipelined: operands for chunk kc+1 are issued before the
    // WMMAs of chunk kc so ds latency overlaps matrix math.
    v8f c0 = zero8, c1 = zero8;
    v16bf a0 = cat16(*(const v8bf*)(qap),  *(const v8bf*)(qap + 16));
    v16bf b0 = cat16(*(const v8bf*)(kbp0), *(const v8bf*)(kbp0 + 16));
    v16bf b1 = cat16(*(const v8bf*)(kbp1), *(const v8bf*)(kbp1 + 16));
#pragma unroll
    for (int kc = 0; kc < 8; ++kc) {
      v16bf na = a0, nb0 = b0, nb1 = b1;
      if (kc < 7) {
        const int o = (kc + 1) * 32;
        na  = cat16(*(const v8bf*)(qap + o),  *(const v8bf*)(qap + o + 16));
        nb0 = cat16(*(const v8bf*)(kbp0 + o), *(const v8bf*)(kbp0 + o + 16));
        nb1 = cat16(*(const v8bf*)(kbp1 + o), *(const v8bf*)(kbp1 + o + 16));
      }
      c0 = wmma_bf16(a0, b0, c0);
      c1 = wmma_bf16(a0, b1, c1);
      a0 = na; b0 = nb0; b1 = nb1;
    }

    // ---- decay matrix + online max/sum rescale -----------------------------
    const float g0 = igp[j0 + nn]      - csp[j0 + nn];
    const float g1 = igp[j0 + 16 + nn] - csp[j0 + 16 + nn];
    float alpha[8];
#pragma unroll
    for (int r = 0; r < 8; ++r) {
      const int irow = rbase + r + 8 * half;
      const float ld0 = ((j0 + nn) <= irow)      ? (csq[r] + g0) : -1e30f;
      const float ld1 = ((j0 + 16 + nn) <= irow) ? (csq[r] + g1) : -1e30f;
      float rm = fmaxf(ld0, ld1);
      rm = fmaxf(rm, __shfl_xor(rm, 1, 32));
      rm = fmaxf(rm, __shfl_xor(rm, 2, 32));
      rm = fmaxf(rm, __shfl_xor(rm, 4, 32));
      rm = fmaxf(rm, __shfl_xor(rm, 8, 32));
      const float mnew = fmaxf(mrun[r], rm);
      const float a    = expf(mrun[r] - mnew);
      mrun[r] = mnew;
      const float C0 = c0[r] * expf(ld0 - mnew);
      const float C1 = c1[r] * expf(ld1 - mnew);
      float rs = C0 + C1;
      rs += __shfl_xor(rs, 1, 32);
      rs += __shfl_xor(rs, 2, 32);
      rs += __shfl_xor(rs, 4, 32);
      rs += __shfl_xor(rs, 8, 32);
      lrun[r]  = lrun[r] * a + rs;
      alpha[r] = a;
      Cbf[(r + 8 * half) * 32 + nn]      = (__bf16)C0;
      Cbf[(r + 8 * half) * 32 + 16 + nn] = (__bf16)C1;
    }
    __builtin_amdgcn_wave_barrier();

    // ---- acc = acc*alpha + C * V  (16 WMMAs along DH, pipelined) ----------
    const v16bf ca = cat16(*(const v8bf*)pc, *(const v8bf*)(pc + 16));
    v16bf vb = cat16(*(const v8bf*)(vtp), *(const v8bf*)(vtp + 16));
#pragma unroll
    for (int t = 0; t < 16; ++t) {
      v16bf nvb = vb;
      if (t < 15) {
        const int o = (t + 1) * 512;   // next 16 columns of DH (16*32 elems)
        nvb = cat16(*(const v8bf*)(vtp + o), *(const v8bf*)(vtp + o + 16));
      }
      v8f a = acc[t];
#pragma unroll
      for (int r = 0; r < 8; ++r) a[r] *= alpha[r];
      acc[t] = wmma_bf16(ca, vb, a);
      vb = nvb;
    }
  }

  // ---- normalizer + per-head layernorm + store -----------------------------
  const float* scp = onscale + nh * DH_;
  float sc[16];
#pragma unroll
  for (int t = 0; t < 16; ++t) sc[t] = scp[t * 16 + nn];

#pragma unroll
  for (int r = 0; r < 8; ++r) {
    const float norm = fmaxf(fabsf(lrun[r]), expf(-mrun[r])) + 1e-6f;
    const float inv  = 1.f / norm;
    float h[16];
    float s1 = 0.f, s2 = 0.f;
#pragma unroll
    for (int t = 0; t < 16; ++t) {
      h[t] = acc[t][r] * inv;
      s1 += h[t];
      s2 += h[t] * h[t];
    }
    s1 += __shfl_xor(s1, 1, 32); s2 += __shfl_xor(s2, 1, 32);
    s1 += __shfl_xor(s1, 2, 32); s2 += __shfl_xor(s2, 2, 32);
    s1 += __shfl_xor(s1, 4, 32); s2 += __shfl_xor(s2, 4, 32);
    s1 += __shfl_xor(s1, 8, 32); s2 += __shfl_xor(s2, 8, 32);
    const float mean = s1 * (1.f / 256.f);
    const float var  = s2 * (1.f / 256.f) - mean * mean;
    const float rstd = rsqrtf(var + 1e-5f);
    const size_t ob =
        ((size_t)b * S_ + (rbase + r + 8 * half)) * H_ + (size_t)nh * DH_;
#pragma unroll
    for (int t = 0; t < 16; ++t)
      out[ob + t * 16 + nn] = (h[t] - mean) * rstd * sc[t];
  }
}

// ---------------------------------------------------------------------------
extern "C" void kernel_launch(void* const* d_in, const int* in_sizes, int n_in,
                              void* d_out, int out_size, void* d_ws,
                              size_t ws_size, hipStream_t stream) {
  (void)in_sizes; (void)n_in; (void)out_size; (void)ws_size;
  const float* q   = (const float*)d_in[0];
  const float* k   = (const float*)d_in[1];
  const float* v   = (const float*)d_in[2];
  const float* igk = (const float*)d_in[3];
  const float* igb = (const float*)d_in[4];
  const float* fgk = (const float*)d_in[5];
  const float* fgb = (const float*)d_in[6];
  const float* osc = (const float*)d_in[7];
  float* out = (float*)d_out;

  float* wsf = (float*)d_ws;
  float* ig  = wsf;                          // B*NH*S floats
  float* cs  = wsf + (size_t)B_ * NH_ * S_;  // B*NH*S floats (logsig -> cumsum)

  gates_kernel<<<B_ * S_, 256, 0, stream>>>(q, k, v, igk, igb, fgk, fgb, ig, cs);
  scan_kernel<<<B_ * NH_, 32, 0, stream>>>(cs);
  flash_kernel<<<B_ * NH_ * (S_ / 16), 32, 0, stream>>>(q, k, v, ig, cs, osc, out);
}
